// GridNetBlock_43739946943049
// MI455X (gfx1250) — matrix operand order repaired and checked by reading, not compiled, not run on hardware
//
#include <hip/hip_runtime.h>
#include <hip/hip_bf16.h>
#include <math.h>

// ---------------------------------------------------------------------------
// Types
// ---------------------------------------------------------------------------
typedef _Float16 half_t;
typedef __attribute__((ext_vector_type(16))) _Float16 v16h;
typedef __attribute__((ext_vector_type(2)))  _Float16 h2;
typedef __attribute__((ext_vector_type(8)))  float    v8f;
typedef __attribute__((ext_vector_type(4)))  int      v4i;

// Problem constants (from reference setup_inputs)
#define Bb 4
#define Cc 48
#define Tt 256
#define Ff 65
#define Hh 192
#define G4 768      // 4*H
#define NHn 4
#define Ee 8
#define Dv 12
#define TF 16640    // T*F
#define M1 66560    // B*T*F == B*F*T
#define NSEQ_I 1024 // intra sequences (B*T), already mult of 16
#define NSEQ_E 260  // inter sequences (B*F)
#define NSEQP_E 272 // inter padded to mult of 16
#define EFPAD 544   // pad(E*F=520) to mult of 32
#define DVFPAD 784  // pad(Dv*F=780) to mult of 16

#if defined(__has_builtin)
#if __has_builtin(__builtin_amdgcn_global_load_async_to_lds_b128)
#define HAVE_ASYNC_LDS 1
#endif
#endif

__device__ __forceinline__ float sigm(float x) { return 1.0f / (1.0f + __expf(-x)); }

// ---------------------------------------------------------------------------
// WMMA fragment loaders (layouts per CDNA5 ISA §7.12.2)
// ---------------------------------------------------------------------------
// A fragment (16x32 f16, row-major source). rowp points at
// &A[(m0 + (lane&15)) * lda + k0]. Lanes 0-15 hold K {0..7,16..23},
// lanes 16-31 hold K {8..15,24..31}; VGPR r packs (kk, kk+1).
__device__ __forceinline__ v16h load_a_rowptr(const half_t* __restrict__ rowp, int lane) {
  int kb = (lane >> 4) << 3;  // 0 or 8
  v16h a;
#pragma unroll
  for (int r = 0; r < 8; ++r) {
    int kk = kb + ((r < 4) ? (2 * r) : (16 + 2 * (r - 4)));
    h2 v = *(const h2*)(rowp + kk);
    a[2 * r] = v[0];
    a[2 * r + 1] = v[1];
  }
  return a;
}

// B fragment (32x16 f16) with B stored [N, ldb] row-major (computing A.B^T).
// rowp points at &B[(n0 + (lane&15)) * ldb + k0].
// Lanes 0-15 hold K {0..7,16..23}, lanes 16-31 hold K {8..15,24..31}; VGPR r packs (kk, kk+1).
__device__ __forceinline__ v16h load_bT_rowptr(const half_t* __restrict__ rowp, int lane) {
  int kb = (lane >> 4) << 3;  // 0 or 8
  v16h b;
#pragma unroll
  for (int r = 0; r < 8; ++r) {
    h2 v = *(const h2*)(rowp + kb + ((r < 4) ? (2 * r) : (16 + 2 * (r - 4))));
    b[2 * r] = v[0];
    b[2 * r + 1] = v[1];
  }
  return b;
}

#define WMMA(a, b, c) __builtin_amdgcn_wmma_f32_16x16x32_f16(false, (a), false, (b), (short)0, (c), false, false)

// ---------------------------------------------------------------------------
// Weight convert f32 -> f16, optional K padding; cvt_t emits B^T layout.
// ---------------------------------------------------------------------------
__global__ void cvt_pad(const float* __restrict__ src, half_t* __restrict__ dst,
                        int rows, int cols, int dcols) {
  int idx = blockIdx.x * 256 + threadIdx.x;
  if (idx >= rows * dcols) return;
  int r = idx / dcols, c = idx % dcols;
  dst[idx] = (c < cols) ? (half_t)src[r * cols + c] : (half_t)0;
}

// src [K, N] f32 -> dst [N, K] f16 (so GEMM B loads are K-contiguous)
__global__ void cvt_t(const float* __restrict__ src, half_t* __restrict__ dst,
                      int N, int K) {
  int idx = blockIdx.x * 256 + threadIdx.x;
  if (idx >= N * K) return;
  int n = idx / K, k = idx % K;
  dst[idx] = (half_t)src[k * N + n];
}

__global__ void fill_zero_h(half_t* __restrict__ p, int n) {
  int i = blockIdx.x * 256 + threadIdx.x;
  if (i < n) p[i] = (half_t)0;
}

// ---------------------------------------------------------------------------
// LayerNorm over channel dim (C=48) of x[B,C,T,F]; emit f16 rows [*, 64]
// modeInter==0: row = (b*T+t)*F + f   (intra: seq along frequency)
// modeInter==1: row = (b*F+f)*T + t   (inter: seq along time)
// ---------------------------------------------------------------------------
__global__ __launch_bounds__(256) void ln_channel(const float* __restrict__ X,
                                                  const float* __restrict__ gamma,
                                                  const float* __restrict__ beta,
                                                  half_t* __restrict__ out, int modeInter) {
  int idx = blockIdx.x * 256 + threadIdx.x;
  if (idx >= Bb * Tt * Ff) return;
  int f = idx % Ff;
  int t = (idx / Ff) & (Tt - 1);
  int b = idx / (Ff * Tt);
  const float* xp = X + (long)b * Cc * TF + t * Ff + f;
  float s = 0.f, s2 = 0.f;
  for (int c = 0; c < Cc; ++c) {
    float v = xp[(long)c * TF];
    s += v;
    s2 += v * v;
  }
  float mu = s * (1.0f / Cc);
  float var = s2 * (1.0f / Cc) - mu * mu;
  float rstd = rsqrtf(var + 1e-5f);
  long row = modeInter ? ((long)(b * Ff + f) * Tt + t) : ((long)(b * Tt + t) * Ff + f);
  half_t* op = out + row * 64;
  for (int c = 0; c < Cc; ++c)
    op[c] = (half_t)((xp[(long)c * TF] - mu) * rstd * gamma[c] + beta[c]);
  for (int c = Cc; c < 64; ++c) op[c] = (half_t)0;
}

// ---------------------------------------------------------------------------
// BiLSTM direction kernel. One block = 16 sequences, 12 waves (384 threads).
// U (768x192 f16 = 288KB) staged in LDS (fits 320KB WGP LDS; async-to-LDS
// path when available), h tile (6KB) in LDS, c in accumulator VGPRs.
// Wave w owns hidden slice [16w,16w+16) and computes its i/f/g/o 16x16 gate
// tiles so the gate nonlinearity is lane-local (no cross-wave traffic).
// Gates = x_t.W^T + h.U^T + b via v_wmma_f32_16x16x32_f16 (K=64pad + K=192).
// Sequence count is padded to a multiple of 16 (zero rows) so all fragment
// loads are unconditional — no exec-mask branching in the recurrence loop.
// blockIdx.y = direction (0 fwd writes cols [0,H), 1 bwd writes [H,2H)).
// ---------------------------------------------------------------------------
__global__ __launch_bounds__(384) void lstm_wmma(const half_t* __restrict__ xi,  // [nseqPad, L, 64]
                                                 const half_t* __restrict__ Wfw,
                                                 const half_t* __restrict__ Wbw,  // [768,64]
                                                 const half_t* __restrict__ Ufw,
                                                 const half_t* __restrict__ Ubw,  // [768,192]
                                                 const float* __restrict__ bfw,
                                                 const float* __restrict__ bbw,  // [768]
                                                 half_t* __restrict__ hout,       // [nseq, L, 384]
                                                 int nseq, int L) {
  extern __shared__ char smem[];
  half_t* Ulds = (half_t*)smem;                                 // 768*192 f16
  half_t* hlds = (half_t*)(smem + G4 * Hh * sizeof(half_t));    // 16*192 f16
  const int dir = blockIdx.y;
  const half_t* W = dir ? Wbw : Wfw;
  const half_t* U = dir ? Ubw : Ufw;
  const float* bias = dir ? bbw : bfw;
  const int seq0 = blockIdx.x * 16;
  const int tid = threadIdx.x, wave = tid >> 5, lane = tid & 31;

#ifdef HAVE_ASYNC_LDS
  // Async DMA of U into LDS: 16B per lane per op, tracked by ASYNCcnt.
  for (int i = tid * 8; i < G4 * Hh; i += 384 * 8)
    __builtin_amdgcn_global_load_async_to_lds_b128(
        (__attribute__((address_space(1))) v4i*)(U + i),
        (__attribute__((address_space(3))) v4i*)(Ulds + i), 0, 0);
  asm volatile("s_wait_asynccnt 0x0" ::: "memory");
#else
  for (int i = tid; i < G4 * Hh / 2; i += 384) ((h2*)Ulds)[i] = ((const h2*)U)[i];
#endif
  for (int i = tid; i < 16 * Hh; i += 384) hlds[i] = (half_t)0;
  __syncthreads();

  const int hcol = (wave << 4) + (lane & 15);  // hidden col 0..191 for this lane
  const float b_i = bias[hcol];
  const float b_f = bias[Hh + hcol];
  const float b_g = bias[2 * Hh + hcol];
  const float b_o = bias[3 * Hh + hcol];
  const int arow = lane & 15;  // local sequence row for A loads
  const half_t* xbase = xi + (long)(seq0 + arow) * L * 64;

  v8f cst = {};  // cell state: (seq = r + 8*(lane>=16), hidden = hcol)
  for (int s = 0; s < L; ++s) {
    const int t = dir ? (L - 1 - s) : s;
    v8f ai, af, ag, ao;
#pragma unroll
    for (int r = 0; r < 8; ++r) { ai[r] = b_i; af[r] = b_f; ag[r] = b_g; ao[r] = b_o; }

    // x contribution: K = 64 (padded 48), W streamed from L2
    const half_t* xrow = xbase + (long)t * 64;
#pragma unroll
    for (int kx = 0; kx < 64; kx += 32) {
      v16h a = load_a_rowptr(xrow + kx, lane);
      v16h w0 = load_bT_rowptr(W + (long)(0 * Hh + hcol) * 64 + kx, lane);
      v16h w1 = load_bT_rowptr(W + (long)(1 * Hh + hcol) * 64 + kx, lane);
      v16h w2 = load_bT_rowptr(W + (long)(2 * Hh + hcol) * 64 + kx, lane);
      v16h w3 = load_bT_rowptr(W + (long)(3 * Hh + hcol) * 64 + kx, lane);
      ai = WMMA(a, w0, ai);
      af = WMMA(a, w1, af);
      ag = WMMA(a, w2, ag);
      ao = WMMA(a, w3, ao);
    }
    // h contribution: K = 192, U from LDS
#pragma unroll
    for (int kh = 0; kh < Hh; kh += 32) {
      v16h a = load_a_rowptr(hlds + arow * Hh + kh, lane);
      v16h u0 = load_bT_rowptr(Ulds + (long)(0 * Hh + hcol) * Hh + kh, lane);
      v16h u1 = load_bT_rowptr(Ulds + (long)(1 * Hh + hcol) * Hh + kh, lane);
      v16h u2 = load_bT_rowptr(Ulds + (long)(2 * Hh + hcol) * Hh + kh, lane);
      v16h u3 = load_bT_rowptr(Ulds + (long)(3 * Hh + hcol) * Hh + kh, lane);
      ai = WMMA(a, u0, ai);
      af = WMMA(a, u1, af);
      ag = WMMA(a, u2, ag);
      ao = WMMA(a, u3, ao);
    }
    __syncthreads();  // all waves finished reading hlds

    // lane-local gate nonlinearity + state update
#pragma unroll
    for (int r = 0; r < 8; ++r) {
      float gi = sigm(ai[r]), gf = sigm(af[r]), gg = tanhf(ag[r]), go = sigm(ao[r]);
      float cn = gf * cst[r] + gi * gg;
      cst[r] = cn;
      float hv = go * tanhf(cn);
      int m = r + ((lane >> 4) << 3);
      hlds[m * Hh + hcol] = (half_t)hv;
      int sq = seq0 + m;
      if (sq < nseq)
        hout[((long)sq * L + t) * (2 * Hh) + dir * Hh + hcol] = (half_t)hv;
    }
    __syncthreads();  // hlds ready for next step
  }
}

// ---------------------------------------------------------------------------
// Generic batched WMMA GEMM: out = epilogue(A * B^T * scale + bias), where
// B is always stored [N, ldb>=K] row-major (K-contiguous fragment loads).
// 8 waves/block, one 16x16 tile per wave. M,N multiples of 16; K mult of 32.
// mode 0: f32 store out[z][m][n]
// mode 2: x1 = x + y (intra decode m=(b*T+t)*F+f, n=c)   [Wli projection]
// mode 3: x2 = x1 + y (inter decode m=(b*F+f)*T+t, n=c)  [Wle projection]
// mode 4: attention-O scatter -> O[b, h*Dv+dv, t, f]
// ---------------------------------------------------------------------------
__global__ __launch_bounds__(256) void gemm_wmma_f16(
    const half_t* __restrict__ A, int lda, long sAz,
    const half_t* __restrict__ Bm, int ldb, long sBz,
    const float* __restrict__ bias,
    float* __restrict__ outF, int ldo, long sOz,
    const float* __restrict__ res,
    int M, int N, int K, float scale, int mode) {
  const int wave = threadIdx.x >> 5, lane = threadIdx.x & 31;
  const int nt = N >> 4;
  const long mt = (long)(M >> 4);
  const long tile = (long)blockIdx.x * 8 + wave;
  if (tile >= mt * nt) return;
  const int tn = (int)(tile % nt);
  const int m0 = (int)(tile / nt) << 4;
  const int n0 = tn << 4;

  v8f acc = {};
  const half_t* arp = A + (long)blockIdx.z * sAz + (long)(m0 + (lane & 15)) * lda;
  const half_t* brp = Bm + (long)blockIdx.z * sBz + (long)(n0 + (lane & 15)) * ldb;
  for (int k0 = 0; k0 < K; k0 += 32) {
    v16h a = load_a_rowptr(arp + k0, lane);
    v16h b = load_bT_rowptr(brp + k0, lane);
    acc = WMMA(a, b, acc);
  }

  const int col = n0 + (lane & 15);
  const int rbase = m0 + ((lane >> 4) << 3);
  const float bv = bias ? bias[col] : 0.0f;
#pragma unroll
  for (int r = 0; r < 8; ++r) {
    int row = rbase + r;
    float v = acc[r] * scale + bv;
    if (mode == 0) {
      outF[(long)blockIdx.z * sOz + (long)row * ldo + col] = v;
    } else if (mode == 2) {
      int f = row % Ff;
      int bt = row / Ff;
      int t = bt & (Tt - 1);
      int b = bt >> 8;
      long idx = ((long)(b * Cc + col) * Tt + t) * Ff + f;
      outF[idx] = res[idx] + v;
    } else if (mode == 3) {
      int t = row & (Tt - 1);
      int sq = row >> 8;
      int f = sq % Ff;
      int b = sq / Ff;
      long idx = ((long)(b * Cc + col) * Tt + t) * Ff + f;
      outF[idx] = res[idx] + v;
    } else {  // mode 4: O[b, h*Dv+dv, t=row, f]
      if (col < Dv * Ff) {
        int z = blockIdx.z;
        int b = z >> 2, h = z & 3;
        int dv = col / Ff, f = col % Ff;
        long idx = ((long)(b * Cc + h * Dv + dv) * Tt + row) * Ff + f;
        outF[idx] = v;
      }
    }
  }
}

// ---------------------------------------------------------------------------
// Per-head 1x1 conv + PReLU:  out[b,h,e,t,f] (f32)
// ---------------------------------------------------------------------------
__global__ __launch_bounds__(256) void proj1x1(const float* __restrict__ X,
                                               const float* __restrict__ W,
                                               const float* __restrict__ bias,
                                               const float* __restrict__ alpha,
                                               float* __restrict__ out, int Eo) {
  int idx = blockIdx.x * 256 + threadIdx.x;
  int total = Bb * NHn * Eo * Tt * Ff;
  if (idx >= total) return;
  int f = idx % Ff;
  int r = idx / Ff;
  int t = r & (Tt - 1); r >>= 8;
  int e = r % Eo; r /= Eo;
  int h = r & 3;
  int b = r >> 2;
  const float* wp = W + (h * Eo + e) * Cc;
  const float* xp = X + (long)b * Cc * TF + t * Ff + f;
  float acc = bias[h * Eo + e];
  for (int c = 0; c < Cc; ++c) acc += wp[c] * xp[(long)c * TF];
  float a = alpha[h];
  out[idx] = acc >= 0.f ? acc : a * acc;
}

// ---------------------------------------------------------------------------
// LN over (E,F) per (b,h,t) + pack f16. transT==0: row-major [z][t][e*F+f]
// (padded). transT==1: K-major [z][e*F+f][t] so it serves as a GEMM B^T
// operand with contiguous K loads (used for V).
// ---------------------------------------------------------------------------
__global__ __launch_bounds__(256) void ln_ef_pack(const float* __restrict__ Y,
                                                  const float* __restrict__ g,
                                                  const float* __restrict__ zb,
                                                  half_t* __restrict__ out, int Eo, int Npad,
                                                  int transT) {
  __shared__ float rs[256], rs2[256];
  const int grp = blockIdx.x;  // (b*4+h)*256 + t
  const int t = grp & (Tt - 1);
  const int h = (grp >> 8) & 3;
  const int b = grp >> 10;
  const int z = grp >> 8;
  const int n = Eo * Ff;
  const float* base = Y + ((long)(b * NHn + h) * Eo * Tt) * Ff + t * Ff;  // +e*TF+f
  float s = 0.f, s2 = 0.f;
  for (int i = threadIdx.x; i < n; i += 256) {
    int e = i / Ff, f = i % Ff;
    float v = base[(long)e * TF + f];
    s += v;
    s2 += v * v;
  }
  rs[threadIdx.x] = s;
  rs2[threadIdx.x] = s2;
  __syncthreads();
  for (int off = 128; off > 0; off >>= 1) {
    if (threadIdx.x < off) {
      rs[threadIdx.x] += rs[threadIdx.x + off];
      rs2[threadIdx.x] += rs2[threadIdx.x + off];
    }
    __syncthreads();
  }
  float mu = rs[0] / n;
  float rstd = rsqrtf(rs2[0] / n - mu * mu + 1e-5f);
  if (transT) {
    half_t* op = out + (long)z * Npad * Tt;
    for (int i = threadIdx.x; i < n; i += 256) {
      int e = i / Ff, f = i % Ff;
      float v = base[(long)e * TF + f];
      int gi = (h * Eo + e) * Ff + f;
      op[(long)i * Tt + t] = (half_t)((v - mu) * rstd * g[gi] + zb[gi]);
    }
    for (int i = n + threadIdx.x; i < Npad; i += 256) op[(long)i * Tt + t] = (half_t)0;
  } else {
    half_t* op = out + (long)grp * Npad;
    for (int i = threadIdx.x; i < n; i += 256) {
      int e = i / Ff, f = i % Ff;
      float v = base[(long)e * TF + f];
      int gi = (h * Eo + e) * Ff + f;
      op[i] = (half_t)((v - mu) * rstd * g[gi] + zb[gi]);
    }
    for (int i = n + threadIdx.x; i < Npad; i += 256) op[i] = (half_t)0;
  }
}

// ---------------------------------------------------------------------------
// Row softmax over 256 logits -> f16 probabilities
// ---------------------------------------------------------------------------
__global__ __launch_bounds__(256) void softmax_rows(const float* __restrict__ Lg,
                                                    half_t* __restrict__ P) {
  __shared__ float red[256];
  const long row = blockIdx.x;
  float v = Lg[row * 256 + threadIdx.x];
  red[threadIdx.x] = v;
  __syncthreads();
  for (int off = 128; off > 0; off >>= 1) {
    if (threadIdx.x < off) red[threadIdx.x] = fmaxf(red[threadIdx.x], red[threadIdx.x + off]);
    __syncthreads();
  }
  float mx = red[0];
  __syncthreads();
  float e = __expf(v - mx);
  red[threadIdx.x] = e;
  __syncthreads();
  for (int off = 128; off > 0; off >>= 1) {
    if (threadIdx.x < off) red[threadIdx.x] += red[threadIdx.x + off];
    __syncthreads();
  }
  P[row * 256 + threadIdx.x] = (half_t)(e / red[0]);
}

// ---------------------------------------------------------------------------
// Final concat projection: y0[b,d,t,f] = PReLU(Wp.O + bp)
// ---------------------------------------------------------------------------
__global__ __launch_bounds__(256) void proj_final(const float* __restrict__ O,
                                                  const float* __restrict__ Wp,
                                                  const float* __restrict__ bp,
                                                  const float* __restrict__ ap,
                                                  float* __restrict__ y0) {
  int idx = blockIdx.x * 256 + threadIdx.x;
  if (idx >= Bb * Cc * Tt * Ff) return;
  int f = idx % Ff;
  int r = idx / Ff;
  int t = r & (Tt - 1); r >>= 8;
  int d = r % Cc;
  int b = r / Cc;
  const float* op = O + (long)b * Cc * TF + t * Ff + f;
  const float* wp = Wp + d * Cc;
  float acc = bp[d];
  for (int c = 0; c < Cc; ++c) acc += wp[c] * op[(long)c * TF];
  float a = ap[0];
  y0[idx] = acc >= 0.f ? acc : a * acc;
}

// ---------------------------------------------------------------------------
// Final LN over (C,F) per (b,t), scale gp/zp, + residual x2 -> d_out
// ---------------------------------------------------------------------------
__global__ __launch_bounds__(256) void ln_cf_final(const float* __restrict__ y0,
                                                   const float* __restrict__ gp,
                                                   const float* __restrict__ zp,
                                                   const float* __restrict__ x2,
                                                   float* __restrict__ out) {
  __shared__ float rs[256], rs2[256];
  const int grp = blockIdx.x;  // b*T + t
  const int t = grp & (Tt - 1);
  const int b = grp >> 8;
  const int n = Cc * Ff;
  const float* base = y0 + (long)b * Cc * TF + t * Ff;  // +c*TF+f
  float s = 0.f, s2 = 0.f;
  for (int i = threadIdx.x; i < n; i += 256) {
    int c = i / Ff, f = i % Ff;
    float v = base[(long)c * TF + f];
    s += v;
    s2 += v * v;
  }
  rs[threadIdx.x] = s;
  rs2[threadIdx.x] = s2;
  __syncthreads();
  for (int off = 128; off > 0; off >>= 1) {
    if (threadIdx.x < off) {
      rs[threadIdx.x] += rs[threadIdx.x + off];
      rs2[threadIdx.x] += rs2[threadIdx.x + off];
    }
    __syncthreads();
  }
  float mu = rs[0] / n;
  float rstd = rsqrtf(rs2[0] / n - mu * mu + 1e-5f);
  for (int i = threadIdx.x; i < n; i += 256) {
    int c = i / Ff, f = i % Ff;
    long idx = ((long)(b * Cc + c) * Tt + t) * Ff + f;
    float v = base[(long)c * TF + f];
    out[idx] = (v - mu) * rstd * gp[c * Ff + f] + zp[c * Ff + f] + x2[idx];
  }
}

// ---------------------------------------------------------------------------
// Host orchestration
// ---------------------------------------------------------------------------
extern "C" void kernel_launch(void* const* d_in, const int* in_sizes, int n_in,
                              void* d_out, int out_size, void* d_ws, size_t ws_size,
                              hipStream_t stream) {
  (void)in_sizes; (void)n_in; (void)out_size; (void)ws_size;
  const float* x       = (const float*)d_in[0];
  const float* g_intra = (const float*)d_in[1];
  const float* b_intra = (const float*)d_in[2];
  const float* g_inter = (const float*)d_in[3];
  const float* b_inter = (const float*)d_in[4];
  const float* ia_Wf = (const float*)d_in[5];
  const float* ia_Uf = (const float*)d_in[6];
  const float* ia_bf = (const float*)d_in[7];
  const float* ia_Wb = (const float*)d_in[8];
  const float* ia_Ub = (const float*)d_in[9];
  const float* ia_bb = (const float*)d_in[10];
  const float* ie_Wf = (const float*)d_in[11];
  const float* ie_Uf = (const float*)d_in[12];
  const float* ie_bf = (const float*)d_in[13];
  const float* ie_Wb = (const float*)d_in[14];
  const float* ie_Ub = (const float*)d_in[15];
  const float* ie_bb = (const float*)d_in[16];
  const float* Wli = (const float*)d_in[17];
  const float* bli = (const float*)d_in[18];
  const float* Wle = (const float*)d_in[19];
  const float* ble = (const float*)d_in[20];
  const float* Wq = (const float*)d_in[21];
  const float* bq = (const float*)d_in[22];
  const float* aq = (const float*)d_in[23];
  const float* gq = (const float*)d_in[24];
  const float* zq = (const float*)d_in[25];
  const float* Wk = (const float*)d_in[26];
  const float* bk = (const float*)d_in[27];
  const float* ak = (const float*)d_in[28];
  const float* gk = (const float*)d_in[29];
  const float* zk = (const float*)d_in[30];
  const float* Wv = (const float*)d_in[31];
  const float* bv = (const float*)d_in[32];
  const float* av = (const float*)d_in[33];
  const float* gv = (const float*)d_in[34];
  const float* zv = (const float*)d_in[35];
  const float* Wp = (const float*)d_in[36];
  const float* bp = (const float*)d_in[37];
  const float* ap = (const float*)d_in[38];
  const float* gp = (const float*)d_in[39];
  const float* zp = (const float*)d_in[40];
  float* outp = (float*)d_out;

  // Workspace carve-out (all offsets deterministic).
  // xi buffer is sized for the larger (padded) inter layout: 272*256 rows.
  char* ws = (char*)d_ws;
  size_t off = 0;
  auto bump = [&](size_t bytes) {
    size_t p = off;
    off += (bytes + 255) & ~(size_t)255;
    return p;
  };
  half_t* xi16 = (half_t*)(ws + bump((size_t)NSEQP_E * Tt * 64 * 2));  // reused as xe16
  half_t* hi16 = (half_t*)(ws + bump((size_t)M1 * 384 * 2));           // reused as he16
  float*  x1   = (float*)(ws + bump((size_t)Bb * Cc * TF * 4));
  float*  x2   = (float*)(ws + bump((size_t)Bb * Cc * TF * 4));
  half_t* WiaF = (half_t*)(ws + bump((size_t)G4 * 64 * 2));
  half_t* WiaB = (half_t*)(ws + bump((size_t)G4 * 64 * 2));
  half_t* WieF = (half_t*)(ws + bump((size_t)G4 * 64 * 2));
  half_t* WieB = (half_t*)(ws + bump((size_t)G4 * 64 * 2));
  half_t* UiaF = (half_t*)(ws + bump((size_t)G4 * Hh * 2));
  half_t* UiaB = (half_t*)(ws + bump((size_t)G4 * Hh * 2));
  half_t* UieF = (half_t*)(ws + bump((size_t)G4 * Hh * 2));
  half_t* UieB = (half_t*)(ws + bump((size_t)G4 * Hh * 2));
  half_t* Wli16 = (half_t*)(ws + bump((size_t)Cc * 384 * 2));  // stored [N=48, K=384]
  half_t* Wle16 = (half_t*)(ws + bump((size_t)Cc * 384 * 2));
  float*  Qt = (float*)(ws + bump((size_t)Bb * NHn * Ee * TF * 4));  // reused as y0
  float*  Kt = (float*)(ws + bump((size_t)Bb * NHn * Ee * TF * 4));
  float*  Vt = (float*)(ws + bump((size_t)Bb * NHn * Dv * TF * 4));
  half_t* Qf = (half_t*)(ws + bump((size_t)16 * Tt * EFPAD * 2));
  half_t* Kf = (half_t*)(ws + bump((size_t)16 * Tt * EFPAD * 2));
  half_t* Vf = (half_t*)(ws + bump((size_t)16 * DVFPAD * Tt * 2));  // K-major [z][n][t]
  float*  logits = (float*)(ws + bump((size_t)16 * Tt * Tt * 4));
  half_t* P16 = (half_t*)(ws + bump((size_t)16 * Tt * Tt * 2));
  float*  Obuf = (float*)(ws + bump((size_t)Bb * Cc * TF * 4));
  half_t* xe16 = xi16;
  half_t* he16 = hi16;
  float*  y0 = Qt;

  const size_t lstm_lds = (size_t)G4 * Hh * sizeof(half_t) + 16 * Hh * sizeof(half_t);  // 301056 B

  // 0) weight conversion to f16 (K-padded / transposed where needed)
  auto cvt = [&](const float* s, half_t* dst, int rows, int cols, int dcols) {
    int tot = rows * dcols;
    cvt_pad<<<(tot + 255) / 256, 256, 0, stream>>>(s, dst, rows, cols, dcols);
  };
  cvt(ia_Wf, WiaF, G4, Cc, 64);  cvt(ia_Wb, WiaB, G4, Cc, 64);
  cvt(ie_Wf, WieF, G4, Cc, 64);  cvt(ie_Wb, WieB, G4, Cc, 64);
  cvt(ia_Uf, UiaF, G4, Hh, Hh);  cvt(ia_Ub, UiaB, G4, Hh, Hh);
  cvt(ie_Uf, UieF, G4, Hh, Hh);  cvt(ie_Ub, UieB, G4, Hh, Hh);
  cvt_t<<<(Cc * 384 + 255) / 256, 256, 0, stream>>>(Wli, Wli16, Cc, 384);
  cvt_t<<<(Cc * 384 + 255) / 256, 256, 0, stream>>>(Wle, Wle16, Cc, 384);

  // 1) intra LN -> xi16 [(b*T+t)*F+f, 64]
  ln_channel<<<M1 / 256, 256, 0, stream>>>(x, g_intra, b_intra, xi16, 0);
  // 2) intra BiLSTM (1024 seqs, 65 steps)
  lstm_wmma<<<dim3(NSEQ_I / 16, 2), 384, lstm_lds, stream>>>(xi16, WiaF, WiaB, UiaF, UiaB,
                                                             ia_bf, ia_bb, hi16, NSEQ_I, Ff);
  // 3) Wli projection + residual -> x1 (WMMA, M=66560 N=48 K=384)
  gemm_wmma_f16<<<1560, 256, 0, stream>>>(hi16, 384, 0, Wli16, 384, 0, bli,
                                          x1, 0, 0, x, M1, 48, 384, 1.0f, 2);
  // 4) inter LN -> xe16 [(b*F+f)*T+t, 64]; zero the 12 padded tail sequences
  fill_zero_h<<<(NSEQP_E - NSEQ_E) * Tt * 64 / 256, 256, 0, stream>>>(
      xe16 + (size_t)NSEQ_E * Tt * 64, (NSEQP_E - NSEQ_E) * Tt * 64);
  ln_channel<<<M1 / 256, 256, 0, stream>>>(x1, g_inter, b_inter, xe16, 1);
  // 5) inter BiLSTM (260 seqs padded to 272, 256 steps)
  lstm_wmma<<<dim3(NSEQP_E / 16, 2), 384, lstm_lds, stream>>>(xe16, WieF, WieB, UieF, UieB,
                                                              ie_bf, ie_bb, he16, NSEQ_E, Tt);
  // 6) Wle projection + residual -> x2
  gemm_wmma_f16<<<1560, 256, 0, stream>>>(he16, 384, 0, Wle16, 384, 0, ble,
                                          x2, 0, 0, x1, M1, 48, 384, 1.0f, 3);
  // 7) QKV 1x1 conv + PReLU
  proj1x1<<<(Bb * NHn * Ee * TF) / 256, 256, 0, stream>>>(x2, Wq, bq, aq, Qt, Ee);
  proj1x1<<<(Bb * NHn * Ee * TF) / 256, 256, 0, stream>>>(x2, Wk, bk, ak, Kt, Ee);
  proj1x1<<<(Bb * NHn * Dv * TF) / 256, 256, 0, stream>>>(x2, Wv, bv, av, Vt, Dv);
  // 8) LN over (E,F) + pack to f16 (V packed K-major for the P.V GEMM)
  ln_ef_pack<<<Bb * NHn * Tt, 256, 0, stream>>>(Qt, gq, zq, Qf, Ee, EFPAD, 0);
  ln_ef_pack<<<Bb * NHn * Tt, 256, 0, stream>>>(Kt, gk, zk, Kf, Ee, EFPAD, 0);
  ln_ef_pack<<<Bb * NHn * Tt, 256, 0, stream>>>(Vt, gv, zv, Vf, Dv, DVFPAD, 1);
  // 9) logits = Q.K^T / sqrt(E*F)  (batched WMMA, z = b*NH+h)
  gemm_wmma_f16<<<dim3(32, 1, 16), 256, 0, stream>>>(Qf, EFPAD, (long)Tt * EFPAD,
                                                     Kf, EFPAD, (long)Tt * EFPAD,
                                                     nullptr, logits, Tt, (long)Tt * Tt,
                                                     nullptr, Tt, Tt, EFPAD,
                                                     1.0f / sqrtf((float)(Ee * Ff)), 0);
  // 10) softmax rows -> P16
  softmax_rows<<<16 * Tt, 256, 0, stream>>>(logits, P16);
  // 11) O = P.V, scattered to [B, NH*Dv, T, F] (batched WMMA; V is K-major)
  gemm_wmma_f16<<<dim3(98, 1, 16), 256, 0, stream>>>(P16, Tt, (long)Tt * Tt,
                                                     Vf, Tt, (long)DVFPAD * Tt,
                                                     nullptr, Obuf, 0, 0,
                                                     nullptr, Tt, DVFPAD, Tt, 1.0f, 4);
  // 12) final 1x1 conv + PReLU -> y0
  proj_final<<<(Bb * Cc * TF) / 256, 256, 0, stream>>>(Obuf, Wp, bp, ap, y0);
  // 13) final LN over (C,F) + residual x2 -> out
  ln_cf_final<<<Bb * Tt, 256, 0, stream>>>(y0, gp, zp, x2, outp);
}
